// PatchQwen3MoeExperts_3341484556619
// MI455X (gfx1250) — compile-verified, matrix operand order checked by reading, not run on hardware
//
#include <hip/hip_runtime.h>
#include <hip/hip_bf16.h>

#define NUM_EXPERTS 16
#define HIDDEN 2048
#define INTER 768
#define TOKENS 2048
#define TOP_K 2
#define MAXE (TOKENS * TOP_K)   // worst-case entries in one expert's list
#define TILE_M 32               // tokens per block (two 16-row WMMA tiles)

typedef __attribute__((ext_vector_type(16))) __bf16 v16bf;
typedef __attribute__((ext_vector_type(8)))  __bf16 v8bf;
typedef __attribute__((ext_vector_type(8)))  float  v8f;
typedef __attribute__((ext_vector_type(4)))  float  v4f;

// ---- WMMA helper: D = A*B + C, bf16 inputs, f32 accumulate --------------
__device__ __forceinline__ v8f wmma_bf16(v16bf a, v16bf b, v8f c) {
    // (neg_a, A, neg_b, B, c_mod, C, reuse_a, reuse_b)
    return __builtin_amdgcn_wmma_f32_16x16x32_bf16(false, a, false, b,
                                                   (short)0, c, false, false);
}

// A operand from LDS (row-major bf16 row). ISA layout for 16-bit A 16x32:
// lane L: M = L&15; khalf = L>>4; elems 0..7  = K[khalf*8 .. +7]
//                                 elems 8..15 = K[16+khalf*8 .. +7]
__device__ __forceinline__ v16bf load_a_lds(const __bf16* __restrict__ row,
                                            int kbase, int khalf) {
    v8bf lo = *(const v8bf*)(row + kbase + khalf * 8);
    v8bf hi = *(const v8bf*)(row + kbase + 16 + khalf * 8);
    return __builtin_shufflevector(lo, hi, 0,1,2,3,4,5,6,7,8,9,10,11,12,13,14,15);
}

// B operand from global fp32 weights (row-major [out,in]); lane's column is
// W row (n0 + lane&15), contiguous along reduction dim -> four b128 loads,
// converted to bf16 in-register (v_cvt_pk_bf16_f32).
__device__ __forceinline__ v16bf load_b_f32(const float* __restrict__ rowptr,
                                            int kbase, int khalf) {
    const float* p = rowptr + kbase + khalf * 8;
    v4f a0 = *(const v4f*)(p);
    v4f a1 = *(const v4f*)(p + 4);
    v4f b0 = *(const v4f*)(p + 16);
    v4f b1 = *(const v4f*)(p + 20);
    v16bf r;
    r[0]  = (__bf16)a0[0]; r[1]  = (__bf16)a0[1]; r[2]  = (__bf16)a0[2]; r[3]  = (__bf16)a0[3];
    r[4]  = (__bf16)a1[0]; r[5]  = (__bf16)a1[1]; r[6]  = (__bf16)a1[2]; r[7]  = (__bf16)a1[3];
    r[8]  = (__bf16)b0[0]; r[9]  = (__bf16)b0[1]; r[10] = (__bf16)b0[2]; r[11] = (__bf16)b0[3];
    r[12] = (__bf16)b1[0]; r[13] = (__bf16)b1[1]; r[14] = (__bf16)b1[2]; r[15] = (__bf16)b1[3];
    return r;
}

// silu(g) = g * sigmoid(g), with fast v_rcp_f32 instead of IEEE divide
__device__ __forceinline__ float fast_silu(float g) {
    return g * __builtin_amdgcn_rcpf(1.0f + __expf(-g));
}

// ---- tiny utility kernels ----------------------------------------------
__global__ void zero_f32_kernel(float* __restrict__ p, int n) {
    int i = blockIdx.x * blockDim.x + threadIdx.x;
    if (i < n) p[i] = 0.0f;
}
__global__ void zero_i32_kernel(int* __restrict__ p, int n) {
    int i = blockIdx.x * blockDim.x + threadIdx.x;
    if (i < n) p[i] = 0;
}

// ---- routing: scatter (token, weight) into per-expert lists -------------
__global__ void moe_route_kernel(const int* __restrict__ topk_idx,
                                 const float* __restrict__ topk_w,
                                 int* __restrict__ counts,
                                 int* __restrict__ lists,
                                 float* __restrict__ wts) {
    int i = blockIdx.x * blockDim.x + threadIdx.x;   // flat [T, K] entry
    if (i >= TOKENS * TOP_K) return;
    int e = topk_idx[i];
    if (e < 0) e = 0;
    if (e >= NUM_EXPERTS) e = NUM_EXPERTS - 1;
    int pos = atomicAdd(&counts[e], 1);
    lists[e * MAXE + pos] = i / TOP_K;               // token id
    wts[e * MAXE + pos]   = topk_w[i];
}

// ---- fused expert FFN ----------------------------------------------------
// LDS plan (132 KB): X-tile [32][2048] bf16 staged once; after GEMM1 the
// SiLU(gate)*up results (kept in registers, 96 VGPRs) overwrite the front of
// the same buffer as the [32][768] bf16 h-tile for GEMM2.
#define XS_STRIDE (HIDDEN + 16)   // bf16 elems, 16B-aligned, bank-skewed
#define HS_STRIDE (INTER + 16)
#define NB1 (INTER / 16 / 8)      // 6 gate/up column-blocks per wave
#define SMEM_BYTES (TILE_M * XS_STRIDE * 2 + TILE_M * 4 + TILE_M * 4)

__global__ __launch_bounds__(256)
void moe_ffn_kernel(const float* __restrict__ X,
                    const float* __restrict__ Wg,
                    const float* __restrict__ Wu,
                    const float* __restrict__ Wd,
                    const int* __restrict__ counts,
                    const int* __restrict__ lists,
                    const float* __restrict__ wts,
                    float* __restrict__ out) {
    const int tile = blockIdx.x;          // x-fastest: tiles of one expert
    const int e    = blockIdx.y;          // stay time-adjacent -> L2 reuse
    const int cnt  = counts[e];
    if (tile * TILE_M >= cnt) return;

    extern __shared__ char smem[];
    __bf16* Xs   = (__bf16*)smem;                        // [TILE_M][XS_STRIDE]
    __bf16* Hs   = (__bf16*)smem;                        // overlays Xs later
    int*    toks = (int*)  (smem + TILE_M * XS_STRIDE * 2);
    float*  cws  = (float*)(toks + TILE_M);

    const int tid = threadIdx.x;
    if (tid < TILE_M) {
        int slot = tile * TILE_M + tid;
        if (slot < cnt) {
            toks[tid] = lists[e * MAXE + slot];
            cws[tid]  = wts[e * MAXE + slot];
        } else {                       // pad rows: token 0 with weight 0
            toks[tid] = 0;
            cws[tid]  = 0.0f;
        }
    }
    __syncthreads();

    // Stage TILE_M token rows of X into LDS as bf16.
    for (int v = tid; v < TILE_M * (HIDDEN / 4); v += 256) {
        int row = v >> 9;                  // HIDDEN/4 == 512 vec4 per row
        int c4  = (v & 511) << 2;
        v4f x = *(const v4f*)(X + (size_t)toks[row] * HIDDEN + c4);
        __bf16* dst = Xs + row * XS_STRIDE + c4;
        dst[0] = (__bf16)x[0]; dst[1] = (__bf16)x[1];
        dst[2] = (__bf16)x[2]; dst[3] = (__bf16)x[3];
    }
    __syncthreads();

    const int wave  = tid >> 5;
    const int lane  = tid & 31;
    const int col   = lane & 15;     // M for A-operand, N for B/C/D
    const int khalf = lane >> 4;

    const __bf16* arow0 = Xs + col * XS_STRIDE;          // rows 0..15
    const __bf16* arow1 = Xs + (16 + col) * XS_STRIDE;   // rows 16..31
    const float*  WgE   = Wg + (size_t)e * INTER * HIDDEN;
    const float*  WuE   = Wu + (size_t)e * INTER * HIDDEN;

    // ---- GEMM1: gate & up (N = INTER, K = HIDDEN); both 16-row A-tiles
    //      share every B load. SiLU results kept in registers.
    float hsave[NB1][2][8];
    #pragma unroll
    for (int j = 0; j < NB1; ++j) {
        const int i0 = (wave + j * 8) * 16;
        const float* grow = WgE + (size_t)(i0 + col) * HIDDEN;
        const float* urow = WuE + (size_t)(i0 + col) * HIDDEN;
        v8f accg0 = {}, accu0 = {}, accg1 = {}, accu1 = {};
        for (int kk = 0; kk < HIDDEN / 32; ++kk) {
            const int kbase = kk * 32;
            v16bf a0 = load_a_lds(arow0, kbase, khalf);
            v16bf a1 = load_a_lds(arow1, kbase, khalf);
            v16bf bg = load_b_f32(grow, kbase, khalf);
            v16bf bu = load_b_f32(urow, kbase, khalf);
            accg0 = wmma_bf16(a0, bg, accg0);
            accg1 = wmma_bf16(a1, bg, accg1);
            accu0 = wmma_bf16(a0, bu, accu0);
            accu1 = wmma_bf16(a1, bu, accu1);
        }
        #pragma unroll
        for (int r = 0; r < 8; ++r) {
            hsave[j][0][r] = fast_silu(accg0[r]) * accu0[r];
            hsave[j][1][r] = fast_silu(accg1[r]) * accu1[r];
        }
    }
    __syncthreads();   // all Xs reads complete before overlay

    // Write h-tile (bf16) into LDS, overlaying the X buffer.
    #pragma unroll
    for (int j = 0; j < NB1; ++j) {
        const int i0 = (wave + j * 8) * 16;
        #pragma unroll
        for (int r = 0; r < 8; ++r) {
            const int m = r + 8 * khalf;
            Hs[m        * HS_STRIDE + i0 + col] = (__bf16)hsave[j][0][r];
            Hs[(16 + m) * HS_STRIDE + i0 + col] = (__bf16)hsave[j][1][r];
        }
    }
    __syncthreads();

    // ---- GEMM2: down projection (N = HIDDEN, K = INTER) + weighted scatter
    const float*  WdE   = Wd + (size_t)e * HIDDEN * INTER;
    const __bf16* hrow0 = Hs + col * HS_STRIDE;
    const __bf16* hrow1 = Hs + (16 + col) * HS_STRIDE;
    for (int nb = wave; nb < HIDDEN / 16; nb += 8) {
        const int h0 = nb * 16;
        const float* drow = WdE + (size_t)(h0 + col) * INTER;
        v8f acc0 = {}, acc1 = {};
        for (int kk = 0; kk < INTER / 32; ++kk) {
            const int kbase = kk * 32;
            v16bf a0 = load_a_lds(hrow0, kbase, khalf);
            v16bf a1 = load_a_lds(hrow1, kbase, khalf);
            v16bf b  = load_b_f32(drow, kbase, khalf);
            acc0 = wmma_bf16(a0, b, acc0);
            acc1 = wmma_bf16(a1, b, acc1);
        }
        #pragma unroll
        for (int r = 0; r < 8; ++r) {
            const int m = r + 8 * khalf;
            atomicAdd(out + (size_t)toks[m]      * HIDDEN + h0 + col,
                      acc0[r] * cws[m]);
            atomicAdd(out + (size_t)toks[16 + m] * HIDDEN + h0 + col,
                      acc1[r] * cws[16 + m]);
        }
    }
}

// ---- host-side launcher -------------------------------------------------
extern "C" void kernel_launch(void* const* d_in, const int* in_sizes, int n_in,
                              void* d_out, int out_size, void* d_ws, size_t ws_size,
                              hipStream_t stream) {
    const float* hidden = (const float*)d_in[0];   // [T, H]
    const int*   tkidx  = (const int*)  d_in[1];   // [T, K]
    const float* tkw    = (const float*)d_in[2];   // [T, K]
    const float* gatep  = (const float*)d_in[3];   // [E, I, H]
    const float* upp    = (const float*)d_in[4];   // [E, I, H]
    const float* downp  = (const float*)d_in[5];   // [E, H, I]
    float* out = (float*)d_out;                    // [T, H]

    // workspace layout
    char* ws = (char*)d_ws;
    int*   counts = (int*)ws;                                    // 16 ints
    int*   lists  = (int*)(ws + 256);                            // 16*MAXE ints
    float* wts    = (float*)(ws + 256 + NUM_EXPERTS * MAXE * 4); // 16*MAXE floats

    // 1) zero routing counters + output accumulator
    zero_i32_kernel<<<1, 64, 0, stream>>>(counts, NUM_EXPERTS);
    {
        const int n = TOKENS * HIDDEN;
        zero_f32_kernel<<<(n + 255) / 256, 256, 0, stream>>>(out, n);
    }

    // 2) build per-expert token lists
    {
        const int n = TOKENS * TOP_K;
        moe_route_kernel<<<(n + 255) / 256, 256, 0, stream>>>(
            tkidx, tkw, counts, lists, wts);
    }

    // 3) fused expert FFN: grid (tile, expert); blocks past the expert's
    //    count exit immediately. x-fastest dispatch keeps one expert's
    //    weights hot in L2.
    dim3 grid(MAXE / TILE_M, NUM_EXPERTS);
    moe_ffn_kernel<<<grid, 256, SMEM_BYTES, stream>>>(
        hidden, gatep, upp, downp, counts, lists, wts, out);
}
